// GRUCell_55619826483522
// MI455X (gfx1250) — matrix operand order repaired and checked by reading, not compiled, not run on hardware
//
#include <hip/hip_runtime.h>
#include <hip/hip_bf16.h>
#include <stdint.h>

#define SEQ   512
#define BATCH 64
#define HID   1024
#define BH    (BATCH * HID)       // 65536
#define NBLK  (HID / 16)          // 64 column blocks
#define KBLK  (HID / 32)          // 32 k blocks of 32
#define FRAG_ELEMS 512            // 32 lanes * 16 halves per B fragment
#define PACKED_ELEMS (6u * NBLK * KBLK * FRAG_ELEMS)   // 6 matrices
#define PACKED_BYTES ((size_t)PACKED_ELEMS * 2)        // 12 MB each (hi / lo)

// LDS budget for the recurrent kernel: 3 gates * 32 kblks * 512 bf16 (hi+lo)
// fragments (192 KB) + fp32 accs (12 KB).
#define LDS_FRAG_ELEMS (3 * KBLK * FRAG_ELEMS)         // 49152 per component
#define LDS_BYTES (LDS_FRAG_ELEMS * 2 * 2 + 3 * BATCH * 16 * 4)

typedef __attribute__((ext_vector_type(16))) __bf16 v16bf;
typedef __attribute__((ext_vector_type(8)))  __bf16 v8bf;
typedef __attribute__((ext_vector_type(8)))  float  v8f;

// ---------------------------------------------------------------------------
// WMMA helper: D = A(16x32 bf16) * B(32x16 bf16) + C (f32 accum)
// ---------------------------------------------------------------------------
__device__ __forceinline__ v8f wmma_bf16(v16bf a, v16bf b, v8f c) {
  return __builtin_amdgcn_wmma_f32_16x16x32_bf16(
      /*neg_a=*/false, a, /*neg_b=*/false, b,
      /*c_mod=*/(short)0, c, /*reuse_a=*/false, /*reuse_b=*/false);
}

__device__ __forceinline__ v16bf cat8(v8bf a, v8bf b) {
  return __builtin_shufflevector(a, b, 0, 1, 2, 3, 4, 5, 6, 7,
                                 8, 9, 10, 11, 12, 13, 14, 15);
}

// ---------------------------------------------------------------------------
// A fragment loader from a pre-split bf16 row-major [rows][HID] matrix.
// ISA 16-bit A layout: lanes 0-15: row=lane, K in [0..7],[16..23];
// lanes 16-31: row=lane-16, K in [8..15],[24..31].
// Caller passes row (= tileRow + (lane&15)) and k0 (= kBase + ((lane>>4)<<3)).
// Pure loads: two 16B chunks per lane, zero VALU.
// ---------------------------------------------------------------------------
__device__ __forceinline__ v16bf load_a_frag16(const __bf16* __restrict__ src,
                                               int row, int k0) {
  const v8bf* p = (const v8bf*)(src + (size_t)row * HID + k0);
  return cat8(p[0], p[2]);   // k0..k0+7, k0+16..k0+23
}

// 16B-granular LDS fragment access (ds_load_b128 / ds_store_b128).
__device__ __forceinline__ v16bf lds_load_frag(const __bf16* p) {
  const v8bf* q = (const v8bf*)p;
  return cat8(q[0], q[1]);
}
__device__ __forceinline__ void lds_store_frag(__bf16* p, v16bf v) {
  v8bf a = __builtin_shufflevector(v, v, 0, 1, 2, 3, 4, 5, 6, 7);
  v8bf b = __builtin_shufflevector(v, v, 8, 9, 10, 11, 12, 13, 14, 15);
  ((v8bf*)p)[0] = a;
  ((v8bf*)p)[1] = b;
}

// ---------------------------------------------------------------------------
// Elementwise fp32 -> (bf16 hi, bf16 lo) split.
// ---------------------------------------------------------------------------
__global__ __launch_bounds__(256) void split_f32_kernel(
    const float* __restrict__ src, __bf16* __restrict__ hi,
    __bf16* __restrict__ lo, unsigned n) {
  unsigned i = blockIdx.x * 256u + threadIdx.x;
  if (i >= n) return;
  float f = src[i];
  __bf16 h = (__bf16)f;
  hi[i] = h;
  lo[i] = (__bf16)(f - (float)h);
}

// ---------------------------------------------------------------------------
// Pack weights into WMMA B-fragment order, bf16 hi/lo split.
// Matrices: 0=W_r[:, :H] 1=W_u[:, :H] 2=W_x   (x-side, used by xproj)
//           3=W_r[:, H:] 4=W_u[:, H:] 5=W_o   (h-side, used by recurrence)
// B-frag element (lane, j): k_local = ((j>>3)<<4) + ((lane>>4)<<3) + (j&7), n_local = lane&15.
// ---------------------------------------------------------------------------
__global__ __launch_bounds__(256) void pack_weights_kernel(
    const float* __restrict__ W_r, const float* __restrict__ W_u,
    const float* __restrict__ W_o, const float* __restrict__ W_x,
    __bf16* __restrict__ phi, __bf16* __restrict__ plo) {
  unsigned tid = blockIdx.x * 256u + threadIdx.x;
  if (tid >= PACKED_ELEMS) return;
  unsigned jj   = tid & 15u;
  unsigned lane = (tid >> 4) & 31u;
  unsigned kblk = (tid >> 9) & 31u;
  unsigned nblk = (tid >> 14) & 63u;
  unsigned mat  = tid >> 20;
  unsigned n = nblk * 16u + (lane & 15u);
  unsigned k = kblk * 32u + ((jj >> 3) << 4) + ((lane >> 4) << 3) + (jj & 7u);
  float w;
  switch (mat) {
    case 0:  w = W_r[(size_t)n * 2 * HID + k];        break;
    case 1:  w = W_u[(size_t)n * 2 * HID + k];        break;
    case 2:  w = W_x[(size_t)n * HID + k];            break;
    case 3:  w = W_r[(size_t)n * 2 * HID + HID + k];  break;
    case 4:  w = W_u[(size_t)n * 2 * HID + HID + k];  break;
    default: w = W_o[(size_t)n * HID + k];            break;
  }
  __bf16 h = (__bf16)w;
  phi[tid] = h;
  plo[tid] = (__bf16)(w - (float)h);
}

// ---------------------------------------------------------------------------
// Precompute x-projections for all timesteps (fully parallel big GEMM):
// xproj[gate][t*B + b][n] = x @ Wgate_x.T + bias   for gate in {r, u, x}
// grid = (M/64, NBLK, 3), block = 128 (4 waves, one 16-row M tile each).
// ---------------------------------------------------------------------------
__global__ __launch_bounds__(128) void xproj_kernel(
    const __bf16* __restrict__ in_hi, const __bf16* __restrict__ in_lo,
    const float* __restrict__ b_r, const float* __restrict__ b_u,
    const float* __restrict__ b_x,
    const __bf16* __restrict__ phi, const __bf16* __restrict__ plo,
    float* __restrict__ xproj) {
  const int tid  = threadIdx.x;
  const int lane = tid & 31;
  const int mt   = tid >> 5;                       // wave = M tile 0..3
  const int mbase = blockIdx.x * 64 + mt * 16;
  const int nblk = blockIdx.y;
  const int gate = blockIdx.z;
  const int row  = mbase + (lane & 15);
  const int koff = (lane >> 4) << 3;
  const size_t bbase =
      ((size_t)gate * NBLK + nblk) * KBLK * FRAG_ELEMS + (size_t)lane * 16;

  v8f acc0 = {}, acc1 = {}, acc2 = {};
  for (int kb = 0; kb < KBLK; ++kb) {
    int k0 = kb * 32 + koff;
    v16bf ahi = load_a_frag16(in_hi, row, k0);
    v16bf alo = load_a_frag16(in_lo, row, k0);
    v16bf bhi = *(const v16bf*)(phi + bbase + (size_t)kb * FRAG_ELEMS);
    v16bf blo = *(const v16bf*)(plo + bbase + (size_t)kb * FRAG_ELEMS);
    acc0 = wmma_bf16(ahi, bhi, acc0);   // hi*hi
    acc1 = wmma_bf16(ahi, blo, acc1);   // hi*lo
    acc2 = wmma_bf16(alo, bhi, acc2);   // lo*hi
  }
  v8f acc = acc0 + acc1 + acc2;

  const float* bias = (gate == 0) ? b_r : ((gate == 1) ? b_u : b_x);
  const int n  = nblk * 16 + (lane & 15);
  const float bn = bias[n];
  const int m0 = mbase + ((lane >> 4) << 3);       // C/D layout rows
#pragma unroll
  for (int i = 0; i < 8; ++i) {
    xproj[(size_t)gate * SEQ * BH + (size_t)(m0 + i) * HID + n] = acc[i] + bn;
  }
}

// ---------------------------------------------------------------------------
// Persistent recurrent kernel. grid = 64 WGs (one 16-col block each),
// block = 384 threads = 12 waves: wave = (gate 0..2) x (M tile 0..3).
// Weight B-fragments for this WG (192 KB) are staged into LDS once, then
// reused from LDS (ds_load_b128) across all 512 steps. h is kept as
// pre-split bf16 hi/lo ping-pong buffers written by the previous step's
// postprocess. Device-wide barrier between steps.
// ---------------------------------------------------------------------------
__global__ __launch_bounds__(384) void gru_recurrent_kernel(
    const float* __restrict__ prev_state,
    const float* __restrict__ b_o,
    const __bf16* __restrict__ phi, const __bf16* __restrict__ plo,
    const float* __restrict__ xproj,
    __bf16* h_hi, __bf16* h_lo,      // ping-pong [2][BATCH][HID]
    float* out,                       // [SEQ][B][H] + final state [B][H]
    unsigned* barrier_cnt) {
  // Keep address-space-inferable pointer chains (no integer casts) so these
  // lower to ds_* instructions, not flat.
  extern __shared__ __bf16 smem[];
  __bf16* bfrag_hi = smem;                                // [3][32][512]
  __bf16* bfrag_lo = smem + LDS_FRAG_ELEMS;
  float*  accs     = (float*)(smem + 2 * LDS_FRAG_ELEMS); // [3][64][16]

  const int nblk = blockIdx.x;                     // 0..63
  const int tid  = threadIdx.x;
  const int lane = tid & 31;
  const int wave = tid >> 5;                       // 0..11
  const int mt   = wave & 3;
  const int gate = wave >> 2;                      // 0=r 1=u 2=o
  const int row  = mt * 16 + (lane & 15);          // batch row for A frag
  const int koff = (lane >> 4) << 3;
  const int nlocal = lane & 15;
  const unsigned ngroups = gridDim.x;

  // ---- stage this WG's weight fragments into LDS (once) ----
  {
    const size_t gbase =
        ((size_t)(3 + gate) * NBLK + nblk) * KBLK * FRAG_ELEMS + (size_t)lane * 16;
#pragma unroll
    for (int i = 0; i < 8; ++i) {
      int kb = mt * 8 + i;                         // 4 waves cover 32 kblks
      size_t g = gbase + (size_t)kb * FRAG_ELEMS;
      size_t l = ((size_t)gate * KBLK + kb) * FRAG_ELEMS + (size_t)lane * 16;
      lds_store_frag(bfrag_hi + l, *(const v16bf*)(phi + g));
      lds_store_frag(bfrag_lo + l, *(const v16bf*)(plo + g));
    }
  }
  __syncthreads();

  __bf16* lhi = bfrag_hi + (size_t)gate * KBLK * FRAG_ELEMS + (size_t)lane * 16;
  __bf16* llo = bfrag_lo + (size_t)gate * KBLK * FRAG_ELEMS + (size_t)lane * 16;

  for (int t = 0; t < SEQ; ++t) {
    const __bf16* ahi_src = h_hi + (size_t)(t & 1) * BH;
    const __bf16* alo_src = h_lo + (size_t)(t & 1) * BH;
    const float*  hcur    = (t == 0) ? prev_state : (out + (size_t)(t - 1) * BH);

    v8f acc0 = {}, acc1 = {}, acc2 = {};
#pragma unroll 2
    for (int kb = 0; kb < KBLK; ++kb) {
      int k0 = kb * 32 + koff;
      v16bf ahi = load_a_frag16(ahi_src, row, k0);
      v16bf alo = load_a_frag16(alo_src, row, k0);
      v16bf bhi = lds_load_frag(lhi + (size_t)kb * FRAG_ELEMS);
      v16bf blo = lds_load_frag(llo + (size_t)kb * FRAG_ELEMS);
      acc0 = wmma_bf16(ahi, bhi, acc0);
      acc1 = wmma_bf16(ahi, blo, acc1);
      acc2 = wmma_bf16(alo, bhi, acc2);
    }
    v8f acc = acc0 + acc1 + acc2;

    const int mrow0 = mt * 16 + ((lane >> 4) << 3);
#pragma unroll
    for (int i = 0; i < 8; ++i)
      accs[((size_t)gate * BATCH + mrow0 + i) * 16 + nlocal] = acc[i];
    __syncthreads();

    // Elementwise gate fusion for this WG's 64 rows x 16 cols.
    __bf16* nhi = h_hi + (size_t)((t + 1) & 1) * BH;
    __bf16* nlo = h_lo + (size_t)((t + 1) & 1) * BH;
    for (int idx = tid; idx < BATCH * 16; idx += 384) {
      int m  = idx >> 4;
      int nl = idx & 15;
      int n  = nblk * 16 + nl;
      size_t xo = (size_t)t * BH + (size_t)m * HID + n;
      float r = accs[(size_t)(0 * BATCH + m) * 16 + nl] + xproj[xo];
      float u = accs[(size_t)(1 * BATCH + m) * 16 + nl] + xproj[(size_t)SEQ * BH + xo];
      float o = accs[(size_t)(2 * BATCH + m) * 16 + nl] + b_o[n];
      r = 1.0f / (1.0f + __expf(-r));
      u = 1.0f / (1.0f + __expf(-u));
      float c = tanhf(o * r + xproj[2 * (size_t)SEQ * BH + xo]);
      float hold = hcur[(size_t)m * HID + n];
      float hnew = (1.0f - u) * c + u * hold;
      size_t ho = (size_t)m * HID + n;
      out[(size_t)t * BH + ho] = hnew;
      __bf16 hb = (__bf16)hnew;
      nhi[ho] = hb;
      nlo[ho] = (__bf16)(hnew - (float)hb);
      if (t == SEQ - 1) out[(size_t)SEQ * BH + ho] = hnew;
    }

    // Device-wide step barrier.
    __threadfence();
    __syncthreads();
    if (tid == 0) {
      __hip_atomic_fetch_add(barrier_cnt, 1u, __ATOMIC_ACQ_REL,
                             __HIP_MEMORY_SCOPE_AGENT);
      unsigned target = (unsigned)(t + 1) * ngroups;
      while (__hip_atomic_load(barrier_cnt, __ATOMIC_RELAXED,
                               __HIP_MEMORY_SCOPE_AGENT) < target) {
        __builtin_amdgcn_s_sleep(2);
      }
    }
    __syncthreads();
    __builtin_amdgcn_fence(__ATOMIC_ACQUIRE, "agent");
  }
}

__global__ void init_counter_kernel(unsigned* c) {
  if (threadIdx.x == 0 && blockIdx.x == 0) *c = 0u;
}

// ---------------------------------------------------------------------------
// Inputs: 0 input, 1 prev_state, 2 W_r, 3 b_r, 4 W_u, 5 b_u,
//         6 W_o, 7 b_o, 8 W_x, 9 b_x   (all fp32)
// Output: hidden_seq [512,64,1024] ++ new_state [64,1024]
// ---------------------------------------------------------------------------
extern "C" void kernel_launch(void* const* d_in, const int* in_sizes, int n_in,
                              void* d_out, int out_size, void* d_ws,
                              size_t ws_size, hipStream_t stream) {
  const float* input      = (const float*)d_in[0];
  const float* prev_state = (const float*)d_in[1];
  const float* W_r        = (const float*)d_in[2];
  const float* b_r        = (const float*)d_in[3];
  const float* W_u        = (const float*)d_in[4];
  const float* b_u        = (const float*)d_in[5];
  const float* W_o        = (const float*)d_in[6];
  const float* b_o        = (const float*)d_in[7];
  const float* W_x        = (const float*)d_in[8];
  const float* b_x        = (const float*)d_in[9];

  char* ws = (char*)d_ws;
  size_t off = 0;
  __bf16*   phi     = (__bf16*)(ws + off); off += PACKED_BYTES;           // 12 MB
  __bf16*   plo     = (__bf16*)(ws + off); off += PACKED_BYTES;           // 12 MB
  unsigned* counter = (unsigned*)(ws + off); off += 256;
  float*    xproj   = (float*)(ws + off); off += (size_t)3 * SEQ * BH * 4; // 402 MB
  __bf16*   in_hi   = (__bf16*)(ws + off); off += (size_t)SEQ * BH * 2;    // 64 MB
  __bf16*   in_lo   = (__bf16*)(ws + off); off += (size_t)SEQ * BH * 2;    // 64 MB
  __bf16*   h_hi    = (__bf16*)(ws + off); off += (size_t)2 * BH * 2;      // 256 KB
  __bf16*   h_lo    = (__bf16*)(ws + off); off += (size_t)2 * BH * 2;      // 256 KB
  float*    out     = (float*)d_out;

  hipFuncSetAttribute((const void*)gru_recurrent_kernel,
                      hipFuncAttributeMaxDynamicSharedMemorySize, LDS_BYTES);

  init_counter_kernel<<<1, 32, 0, stream>>>(counter);

  pack_weights_kernel<<<PACKED_ELEMS / 256, 256, 0, stream>>>(
      W_r, W_u, W_o, W_x, phi, plo);

  // Pre-split input and prev_state into bf16 hi/lo.
  split_f32_kernel<<<(SEQ * BH + 255) / 256, 256, 0, stream>>>(
      input, in_hi, in_lo, SEQ * BH);
  split_f32_kernel<<<(BH + 255) / 256, 256, 0, stream>>>(
      prev_state, h_hi, h_lo, BH);   // parity-0 buffer for t = 0

  xproj_kernel<<<dim3(SEQ * BATCH / 64, NBLK, 3), 128, 0, stream>>>(
      in_hi, in_lo, b_r, b_u, b_x, phi, plo, xproj);

  gru_recurrent_kernel<<<NBLK, 384, LDS_BYTES, stream>>>(
      prev_state, b_o, phi, plo, xproj, h_hi, h_lo, out, counter);
}
